// MyBidirectionalLSTM_28269474743011
// MI455X (gfx1250) — compile-verified, matrix operand order checked by reading, not run on hardware
//
#include <hip/hip_runtime.h>
#include <hip/hip_bf16.h>

// ---------------------------------------------------------------------------
// Bidirectional LSTM, B=32, T=512, D=512, H=1024 — CDNA5 (gfx1250) WMMA bf16.
//
// Strategy:
//  * Pack weights [W_hh | W_ih] (K = 1024+512 = 1536) and x (time-major) into
//    the native v_wmma_f32_16x16x32_bf16 fragment layout once per call.
//  * 512 per-timestep kernels (graph-captured): each wave computes one 16x16
//    gate tile with a 48-chunk WMMA K-loop; the block holds all four gates of
//    a hidden slice, so the LSTM cell update is done in LDS in the same kernel.
//  * h state ping-pongs between two packed bf16 buffers (no intra-step races).
// ---------------------------------------------------------------------------

typedef __attribute__((ext_vector_type(16))) __bf16 v16bf;
typedef __attribute__((ext_vector_type(8)))  float  v8f;

#define LSTM_B   32
#define LSTM_T   512
#define LSTM_D   512
#define LSTM_H   1024
#define G4H      4096            // 4*H
#define KTOT     48              // K chunks of 32: 32 (h·W_hh) + 16 (x·W_ih)

// ---- workspace layout (bytes) ----
#define WPACK_OFF  0                         // bf16: 2*256*48*512   = 25165824 B
#define XT_OFF     25165824                  // bf16: 512*2*16*512   = 16777216 B
#define HB0_OFF    41943040                  // bf16: 2*2*32*512     =   131072 B
#define HB1_OFF    42074112                  // bf16:                =   131072 B
#define CB_OFF     42205184                  // f32 : 2*32*1024      =   262144 B
#define BIAS_OFF   42467328                  // f32 : 2*4096         =    32768 B
// total ~42.5 MB

// --- fragment index helpers (ISA 7.12.2, 16-bit A/B 16x32 layout) ---
// lane L holds row/col rc = L&15; K pattern: kbase = (L>>4)*8,
// elements 0..7 -> K = kbase+0..7, elements 8..15 -> K = kbase+16..23.
__device__ __forceinline__ int frag_lane(int rc, int kk) {
    return rc | (((kk >> 3) & 1) << 4);
}
__device__ __forceinline__ int frag_elem(int kk) {
    return (kk & 7) | (((kk >> 4) & 1) << 3);
}

__device__ __forceinline__ unsigned short f2bf(float f) {
    unsigned int u = __float_as_uint(f);
    unsigned int r = u + 0x7FFFu + ((u >> 16) & 1u);   // round-to-nearest-even
    return (unsigned short)(r >> 16);
}

__device__ __forceinline__ float sigmoidf_fast(float x) {
    return 1.0f / (1.0f + __expf(-x));
}
__device__ __forceinline__ float tanhf_fast(float x) {
    float q = __expf(-2.0f * x);
    return (1.0f - q) / (1.0f + q);
}

// ---------------------------------------------------------------------------
// Pack weights: wpack[dir][ntile(256)][kc(48)] = 16x32 bf16 B-fragment.
// kc<32 -> W_hh[g, kc*32+kk] ; kc>=32 -> W_ih[g, (kc-32)*32+kk]
// ---------------------------------------------------------------------------
__global__ __launch_bounds__(256) void k_pack_weights(
    const float* __restrict__ Wf_hh, const float* __restrict__ Wf_ih,
    const float* __restrict__ Wb_hh, const float* __restrict__ Wb_ih,
    unsigned short* __restrict__ wpack)
{
    int id = blockIdx.x * 256 + threadIdx.x;           // < 2*256*48*512
    int kk    = id & 31;
    int rc    = (id >> 5) & 15;
    int kc    = (id >> 9) % 48;
    int rest  = (id >> 9) / 48;
    int ntile = rest & 255;
    int dir   = rest >> 8;
    if (dir >= 2) return;

    int g = ntile * 16 + rc;                           // gate row in [0,4096)
    float v;
    if (kc < 32) {
        const float* Whh = dir ? Wb_hh : Wf_hh;
        v = Whh[g * LSTM_H + kc * 32 + kk];
    } else {
        const float* Wih = dir ? Wb_ih : Wf_ih;
        v = Wih[g * LSTM_D + (kc - 32) * 32 + kk];
    }
    int dst = (((dir * 256 + ntile) * KTOT + kc) << 9)
            + (frag_lane(rc, kk) << 4) + frag_elem(kk);
    wpack[dst] = f2bf(v);
}

// ---------------------------------------------------------------------------
// Pack x time-major: xtpack[t][mt(2)][kc(16)] = 16x32 bf16 A-fragment of
// x[b = mt*16+rc, t, d = kc*32+kk].
// ---------------------------------------------------------------------------
__global__ __launch_bounds__(256) void k_pack_x(
    const float* __restrict__ x, unsigned short* __restrict__ xtpack)
{
    int id = blockIdx.x * 256 + threadIdx.x;           // < 512*2*16*512 = 2^23
    int kk = id & 31;
    int rc = (id >> 5) & 15;
    int kc = (id >> 9) & 15;
    int mt = (id >> 13) & 1;
    int t  = id >> 14;
    if (t >= LSTM_T) return;

    int b = mt * 16 + rc;
    int d = kc * 32 + kk;
    float v = x[(b * LSTM_T + t) * LSTM_D + d];
    int dst = (((t * 2 + mt) * 16 + kc) << 9)
            + (frag_lane(rc, kk) << 4) + frag_elem(kk);
    xtpack[dst] = f2bf(v);
}

// ---------------------------------------------------------------------------
// bias[dir][g] = b_ih[g] + b_hh[g]
// ---------------------------------------------------------------------------
__global__ __launch_bounds__(256) void k_bias(
    const float* __restrict__ bf_ih, const float* __restrict__ bf_hh,
    const float* __restrict__ bb_ih, const float* __restrict__ bb_hh,
    float* __restrict__ bias)
{
    int id = blockIdx.x * 256 + threadIdx.x;           // < 8192
    if (id >= 2 * G4H) return;
    int dir = id >> 12, g = id & (G4H - 1);
    bias[id] = dir ? (bb_ih[g] + bb_hh[g]) : (bf_ih[g] + bf_hh[g]);
}

// ---------------------------------------------------------------------------
// zero h0 fragment buffer and cell state
// ---------------------------------------------------------------------------
__global__ __launch_bounds__(256) void k_init_state(
    unsigned short* __restrict__ h0, float* __restrict__ cbuf)
{
    int id = blockIdx.x * 256 + threadIdx.x;           // < 65536
    if (id < 65536) { h0[id] = 0; cbuf[id] = 0.0f; }
}

// ---------------------------------------------------------------------------
// One LSTM timestep (both directions).
// grid = 128 blocks x 256 threads (8 waves).
//   block: dir = bx>>6 ; mt = (bx>>5)&1 ; htp = bx&31 (pair of hidden tiles)
//   wave w: gate = w&3 (i,f,g,o) ; hsub = w>>2 ; ntile = gate*64 + htp*2+hsub
// Each wave: 48-chunk WMMA K-loop (h fragments then x_t fragments), then the
// block exchanges gate tiles via LDS and performs the cell update.
// ---------------------------------------------------------------------------
__global__ __launch_bounds__(256) void k_lstm_step(
    const unsigned short* __restrict__ wpack,
    const unsigned short* __restrict__ xtpack,
    const unsigned short* __restrict__ h_in,
    unsigned short* __restrict__ h_out,
    float* __restrict__ cbuf,
    const float* __restrict__ bias,
    float* __restrict__ out,
    int t)
{
    const int tid  = threadIdx.x;
    const int lane = tid & 31;
    const int w    = tid >> 5;
    const int dir  = blockIdx.x >> 6;
    const int mt   = (blockIdx.x >> 5) & 1;
    const int htp  = blockIdx.x & 31;
    const int gate = w & 3;
    const int hsub = w >> 2;
    const int ntile = gate * 64 + htp * 2 + hsub;
    const int teff  = dir ? (LSTM_T - 1 - t) : t;      // time index into x / out

    const unsigned short* bptr = wpack  + (((dir * 256 + ntile) * KTOT) << 9) + (lane << 4);
    const unsigned short* hptr = h_in   + (((dir * 2 + mt) * 32) << 9)        + (lane << 4);
    const unsigned short* xptr = xtpack + (((teff * 2 + mt) * 16) << 9)       + (lane << 4);

    v8f acc = {};
    // --- recurrent part: h_{t-1} (K = 0..1023) ---
#pragma unroll 8
    for (int kc = 0; kc < 32; ++kc) {
        v16bf a = *(const v16bf*)(hptr + (kc << 9));
        v16bf b = *(const v16bf*)(bptr + (kc << 9));
        acc = __builtin_amdgcn_wmma_f32_16x16x32_bf16(false, a, false, b,
                                                      (short)0, acc, false, false);
    }
    // --- input part: x_t (K = 1024..1535) ---
#pragma unroll 8
    for (int kc = 0; kc < 16; ++kc) {
        v16bf a = *(const v16bf*)(xptr + (kc << 9));
        v16bf b = *(const v16bf*)(bptr + ((32 + kc) << 9));
        acc = __builtin_amdgcn_wmma_f32_16x16x32_bf16(false, a, false, b,
                                                      (short)0, acc, false, false);
    }

    // --- spill gate tile to LDS (C layout: vgpr r, lane -> row,col) ---
    __shared__ float g_lds[8][16][17];                 // +1 pad: no bank conflicts
    const int col   = lane & 15;
    const int rbase = (lane >> 4) << 3;
    const float bb  = bias[dir * G4H + ntile * 16 + col];
#pragma unroll
    for (int r = 0; r < 8; ++r)
        g_lds[w][rbase + r][col] = acc[r] + bb;
    __syncthreads();

    // --- cell update: 16 batches x 32 hidden columns per block ---
    for (int e = tid; e < 512; e += 256) {
        int m  = e >> 5;            // batch row within tile
        int hc = e & 31;            // hidden col within the pair of tiles
        int hs = hc >> 4;
        int cc = hc & 15;
        float iv = sigmoidf_fast(g_lds[hs * 4 + 0][m][cc]);
        float fv = sigmoidf_fast(g_lds[hs * 4 + 1][m][cc]);
        float gv = tanhf_fast   (g_lds[hs * 4 + 2][m][cc]);
        float ov = sigmoidf_fast(g_lds[hs * 4 + 3][m][cc]);

        int b  = mt * 16 + m;
        int hh = htp * 32 + hc;
        int ci = ((dir << 5) + b) * LSTM_H + hh;
        float cnew = fv * cbuf[ci] + iv * gv;
        cbuf[ci] = cnew;
        float hval = ov * tanhf_fast(cnew);

        // sequence output [B, T, 2H]
        out[(size_t)(b * LSTM_T + teff) * (2 * LSTM_H) + dir * LSTM_H + hh] = hval;
        if (t == LSTM_T - 1) {      // final h, c : [B, 2, H] each
            size_t base = (size_t)LSTM_B * LSTM_T * (2 * LSTM_H);
            out[base + (b * 2 + dir) * LSTM_H + hh] = hval;
            out[base + LSTM_B * 2 * LSTM_H + (b * 2 + dir) * LSTM_H + hh] = cnew;
        }
        // re-pack h into bf16 A-fragment layout for next step
        int kc = hh >> 5, kk = hh & 31;
        h_out[(((dir * 2 + mt) * 32 + kc) << 9)
              + (frag_lane(m, kk) << 4) + frag_elem(kk)] = f2bf(hval);
    }
}

// ---------------------------------------------------------------------------
extern "C" void kernel_launch(void* const* d_in, const int* in_sizes, int n_in,
                              void* d_out, int out_size, void* d_ws, size_t ws_size,
                              hipStream_t stream)
{
    const float* x     = (const float*)d_in[0];
    const float* Wf_ih = (const float*)d_in[1];
    const float* Wf_hh = (const float*)d_in[2];
    const float* bf_ih = (const float*)d_in[3];
    const float* bf_hh = (const float*)d_in[4];
    const float* Wb_ih = (const float*)d_in[5];
    const float* Wb_hh = (const float*)d_in[6];
    const float* bb_ih = (const float*)d_in[7];
    const float* bb_hh = (const float*)d_in[8];

    char* ws = (char*)d_ws;
    unsigned short* wpack  = (unsigned short*)(ws + WPACK_OFF);
    unsigned short* xtpack = (unsigned short*)(ws + XT_OFF);
    unsigned short* hb0    = (unsigned short*)(ws + HB0_OFF);
    unsigned short* hb1    = (unsigned short*)(ws + HB1_OFF);
    float*          cbuf   = (float*)(ws + CB_OFF);
    float*          bias   = (float*)(ws + BIAS_OFF);
    float*          out    = (float*)d_out;

    // one-time (per call) packing + init
    {
        int n = 2 * 256 * KTOT * 512;                   // 12,582,912
        k_pack_weights<<<(n + 255) / 256, 256, 0, stream>>>(Wf_hh, Wf_ih, Wb_hh, Wb_ih, wpack);
    }
    {
        int n = LSTM_T * 2 * 16 * 512;                  // 8,388,608
        k_pack_x<<<(n + 255) / 256, 256, 0, stream>>>(x, xtpack);
    }
    k_bias<<<(2 * G4H + 255) / 256, 256, 0, stream>>>(bf_ih, bf_hh, bb_ih, bb_hh, bias);
    k_init_state<<<65536 / 256, 256, 0, stream>>>(hb0, cbuf);

    // 512 recurrence steps, ping-pong packed-h buffers
    for (int t = 0; t < LSTM_T; ++t) {
        unsigned short* hin  = (t & 1) ? hb1 : hb0;
        unsigned short* hout = (t & 1) ? hb0 : hb1;
        k_lstm_step<<<128, 256, 0, stream>>>(wpack, xtpack, hin, hout, cbuf, bias, out, t);
    }
}
